// LearnableAdjHeteroConv_43550968382024
// MI455X (gfx1250) — matrix operand order repaired and checked by reading, not compiled, not run on hardware
//
#include <hip/hip_runtime.h>

// ---------------------------------------------------------------------------
// MI455X (gfx1250) implementation.
// Pipeline (all GEMMs via v_wmma_f32_16x16x32_bf16, f32 accumulate; global->LDS
// staging via gfx1250 async-to-LDS DMA with double buffering):
//   prep:  W1t, W2t, WfT, WlCatT(/3), WrSumT(/3), blmean(/3), s2=colsum(W2)  (bf16)
//   xT   = transpose(x) bf16                      [B][1024 f][256 d]
//   H    = W1t @ xT^T + b1                        [B][256 n][1024 f] bf16
//   mucat= slice means of H                       [B][3072] bf16
//   alpha= mucat @ WlCatT^T + blmean              [3][64][512] f32
//   Z    = relu(H @ WrSumT^T + alpha_t)           [B][256][512] bf16
//   Zt   = transpose(Z)                           [B][512][256] bf16
//   G    = W2t @ Zt^T                             [B][256 d][512 h] bf16
//   y    = G @ WfT^T + s2[d]*bf[f] + b2[d]        [B][256][1024] f32 -> d_out
// ---------------------------------------------------------------------------

typedef unsigned short u16;
typedef __attribute__((ext_vector_type(8)))  u16    u16x8;
typedef __attribute__((ext_vector_type(8)))  float  v8f;
typedef __attribute__((ext_vector_type(16))) __bf16 bf16x16;

__device__ const int EMAP[9] = {6, 1, 3,   0, 7, 5,   2, 4, 8}; // EMAP[t*3+s]

__device__ __forceinline__ u16 f2bf(float f) {
    unsigned u = __float_as_uint(f);
    unsigned r = u + 0x7FFFu + ((u >> 16) & 1u);   // round-to-nearest-even
    return (u16)(r >> 16);
}
__device__ __forceinline__ float bf2f(u16 h) {
    return __uint_as_float(((unsigned)h) << 16);
}
__device__ __forceinline__ u16 toBF(float f) { return f2bf(f); }
__device__ __forceinline__ u16 toBF(u16 h)   { return h; }

// gfx1250 async global->LDS copy (ASYNCcnt-tracked, per ISA 08_async_tensor).
// LDS byte address = low 32 bits of the generic shared-space pointer
// (ISA: generic LDS addr[31:0] is the wave-relative LDS offset).
__device__ __forceinline__ void async_ld_b128(unsigned lds_off, const u16* gaddr) {
    asm volatile("global_load_async_to_lds_b128 %0, %1, off"
                 :: "v"(lds_off), "v"(gaddr) : "memory");
}
template <int N>
__device__ __forceinline__ void wait_async() {
    if (N == 0)      asm volatile("s_wait_asynccnt 0x0" ::: "memory");
    else if (N == 3) asm volatile("s_wait_asynccnt 0x3" ::: "memory");
    else if (N == 4) asm volatile("s_wait_asynccnt 0x4" ::: "memory");
    else             asm volatile("s_wait_asynccnt 0x0" ::: "memory");
}

// --------------------------- tiny prep kernels -----------------------------

// out[c][r] = bf16(in[r][c])   (small weight transposes)
__global__ void k_transpose_w(const float* __restrict__ in, u16* __restrict__ out,
                              int R, int C) {
    int i = blockIdx.x * 256 + threadIdx.x;
    if (i >= R * C) return;
    int r = i / C, c = i % C;
    out[(size_t)c * R + r] = f2bf(in[i]);
}

// WlCatT[t][h][s*1024+f] = Wl[EMAP[t][s]][f][h] / 3
__global__ void k_wlcat(const float* __restrict__ Wl, u16* __restrict__ out, int total) {
    int i = blockIdx.x * 256 + threadIdx.x;
    if (i >= total) return;
    int t = i / (512 * 3072), rem = i % (512 * 3072);
    int h = rem / 3072, k = rem % 3072;
    int s = k >> 10, f = k & 1023;
    int e = EMAP[t * 3 + s];
    out[i] = f2bf(Wl[((size_t)e * 1024 + f) * 512 + h] * (1.f / 3.f));
}

// WrSumT[t][h][f] = sum_s Wr[EMAP[t][s]][f][h] / 3
__global__ void k_wrsum(const float* __restrict__ Wr, u16* __restrict__ out, int total) {
    int i = blockIdx.x * 256 + threadIdx.x;
    if (i >= total) return;
    int t = i / (512 * 1024), rem = i % (512 * 1024);
    int h = rem / 1024, f = rem % 1024;
    float s = 0.f;
    #pragma unroll
    for (int j = 0; j < 3; j++)
        s += Wr[((size_t)EMAP[t * 3 + j] * 1024 + f) * 512 + h];
    out[i] = f2bf(s * (1.f / 3.f));
}

// blmean[t][h] = mean of 3 bl rows;  s2[d] = sum_n W2[n][d]
__global__ void k_blmean_s2(const float* __restrict__ bl, const float* __restrict__ W2,
                            float* __restrict__ blmean, float* __restrict__ s2) {
    int i = blockIdx.x * 256 + threadIdx.x;
    if (i < 1536) {
        int t = i / 512, h = i % 512;
        float s = 0.f;
        #pragma unroll
        for (int j = 0; j < 3; j++) s += bl[EMAP[t * 3 + j] * 512 + h];
        blmean[i] = s * (1.f / 3.f);
    } else if (i < 1792) {
        int d = i - 1536;
        float s = 0.f;
        for (int n = 0; n < 256; n++) s += W2[n * 256 + d];
        s2[d] = s;
    }
}

// ----------------------- tiled transpose (to bf16) -------------------------
// per z: in [R][C] -> out [C][R] (bf16); grid (C/64, R/64, Z), 256 threads
template <typename TIN>
__global__ __launch_bounds__(256) void k_transpose_tile(const TIN* __restrict__ in,
                                                        u16* __restrict__ out,
                                                        int R, int C) {
    __shared__ u16 t[64 * 80];
    size_t zoff = (size_t)blockIdx.z * R * C;
    int r0 = blockIdx.y * 64, c0 = blockIdx.x * 64;
    for (int i = threadIdx.x; i < 64 * 16; i += 256) {
        int r = i >> 4, cc = (i & 15) << 2;
        const TIN* p = in + zoff + (size_t)(r0 + r) * C + c0 + cc;
        #pragma unroll
        for (int j = 0; j < 4; j++) t[(cc + j) * 80 + r] = toBF(p[j]);
    }
    __syncthreads();
    for (int i = threadIdx.x; i < 64 * 8; i += 256) {
        int cr = i >> 3, rc = (i & 7) << 3;
        u16x8 v = *(const u16x8*)&t[cr * 80 + rc];
        *(u16x8*)(out + zoff + (size_t)(c0 + cr) * R + r0 + rc) = v;
    }
}

// ------------------------------ mu reduce ----------------------------------
// mucat[b][s*1024+f] = mean over node slice s of H[b][n][f]; grid (3, 64)
__global__ __launch_bounds__(256) void k_mu(const u16* __restrict__ H,
                                            u16* __restrict__ mucat) {
    int s = blockIdx.x, b = blockIdx.y;
    int lo  = (s == 0) ? 0 : (s == 1 ? 128 : 192);
    int cnt = (s == 0) ? 128 : 64;
    const u16* h = H + (size_t)b * 256 * 1024;
    float inv = 1.f / (float)cnt;
    for (int f = threadIdx.x; f < 1024; f += 256) {
        float acc = 0.f;
        for (int n = 0; n < cnt; n++) acc += bf2f(h[(size_t)(lo + n) * 1024 + f]);
        mucat[(size_t)b * 3072 + s * 1024 + f] = f2bf(acc * inv);
    }
}

// --------------------------- generic WMMA GEMM -----------------------------
// C[M][N] = A[M][K] * Bt[N][K]^T (+epilogue).  256 threads = 8 waves (wave32).
// Double-buffered LDS tiles filled by global_load_async_to_lds_b128; last
// K-tile peeled so the steady-state loop is branch-free.
// EPI: 0 plain->bf16 | 1 +p0[row]->bf16 | 2 +p1[z*N+col]->f32
//      3 +p0[(t*64+z)*N+col], relu ->bf16 | 4 +p0[row]*p1[col]+p2[row]->f32
// TSEL: select Bt slab + t by node-type of the M-tile (SAGE step).
#define LDK 40  // padded LDS row stride in u16 (80B, 16B aligned)

template <int BM, int BN, int WROWS, int WCOLS, int WTM, int WTN, int EPI, bool TSEL>
__global__ __launch_bounds__(256) void k_gemm(
    const u16* __restrict__ A, const u16* __restrict__ Bt, void* __restrict__ Cv,
    int M, int N, int K,
    long long Abs, long long Bbs, long long Cbs,
    const float* __restrict__ p0, const float* __restrict__ p1,
    const float* __restrict__ p2) {
    constexpr int CA = (BM * 4) / 256;   // 16B chunks per thread for A tile
    constexpr int CB = (BN * 4) / 256;   // 16B chunks per thread for B tile
    constexpr int NPEND = CA + CB;       // async ops issued per thread per tile
    constexpr unsigned BUFA = (unsigned)(BM * LDK * 2);  // bytes per A buffer
    constexpr unsigned BUFB = (unsigned)(BN * LDK * 2);  // bytes per B buffer

    __shared__ u16 sA[2][BM * LDK];
    __shared__ u16 sB[2][BN * LDK];

    int bz = blockIdx.z;
    int m0 = blockIdx.y * BM;
    int n0 = blockIdx.x * BN;
    int t = 0;
    if (TSEL) t = (m0 < 128) ? 0 : (m0 < 192 ? 1 : 2);

    const u16* Ab = A + (size_t)bz * (size_t)Abs;
    const u16* Bb = Bt + (TSEL ? (size_t)t * (size_t)Bbs : (size_t)bz * (size_t)Bbs);

    int tid = threadIdx.x;
    int lane = tid & 31;
    int wave = tid >> 5;
    int wr = wave % WROWS, wc = wave / WROWS;
    int wM = wr * WTM * 16;
    int wN = wc * WTN * 16;

    // per-thread chunk coords + advancing global pointers (step 32 elems/tile)
    const u16* ga[CA];
    unsigned   laA[CA];
    #pragma unroll
    for (int c = 0; c < CA; c++) {
        int i = tid + c * 256;
        int r = i >> 2, kc = (i & 3) << 3;
        ga[c]  = Ab + (size_t)(m0 + r) * K + kc;
        laA[c] = (unsigned)(size_t)&sA[0][r * LDK + kc];
    }
    const u16* gb[CB];
    unsigned   laB[CB];
    #pragma unroll
    for (int c = 0; c < CB; c++) {
        int i = tid + c * 256;
        int r = i >> 2, kc = (i & 3) << 3;
        gb[c]  = Bb + (size_t)(n0 + r) * K + kc;
        laB[c] = (unsigned)(size_t)&sB[0][r * LDK + kc];
    }

    auto issue_tile = [&](int buf) {
        #pragma unroll
        for (int c = 0; c < CA; c++) {
            async_ld_b128(laA[c] + (unsigned)buf * BUFA, ga[c]);
            ga[c] += 32;
        }
        #pragma unroll
        for (int c = 0; c < CB; c++) {
            async_ld_b128(laB[c] + (unsigned)buf * BUFB, gb[c]);
            gb[c] += 32;
        }
    };

    v8f acc[WTM][WTN];
    const v8f vzero = {0.f, 0.f, 0.f, 0.f, 0.f, 0.f, 0.f, 0.f};
    #pragma unroll
    for (int i = 0; i < WTM; i++)
        #pragma unroll
        for (int j = 0; j < WTN; j++) acc[i][j] = vzero;

    int rA  = lane & 15;
    int ksA = (lane >> 4) << 3;   // A frag: K 0-7/16-23 vs 8-15/24-31
    int kh  = (lane >> 4) << 4;   // (unused; B frag uses same K order as A via ksA)

    auto compute_tile = [&](int buf) {
        bf16x16 af[WTM], bfr[WTN];
        #pragma unroll
        for (int i = 0; i < WTM; i++) {
            const u16* p = &sA[buf][(wM + i * 16 + rA) * LDK];
            union { u16x8 h[2]; bf16x16 v; } u;
            u.h[0] = *(const u16x8*)(p + ksA);
            u.h[1] = *(const u16x8*)(p + 16 + ksA);
            af[i] = u.v;
        }
        #pragma unroll
        for (int j = 0; j < WTN; j++) {
            const u16* p = &sB[buf][(wN + j * 16 + rA) * LDK + ksA];
            union { u16x8 h[2]; bf16x16 v; } u;
            u.h[0] = *(const u16x8*)(p);
            u.h[1] = *(const u16x8*)(p + 16);
            bfr[j] = u.v;
        }
        #pragma unroll
        for (int i = 0; i < WTM; i++)
            #pragma unroll
            for (int j = 0; j < WTN; j++)
                acc[i][j] = __builtin_amdgcn_wmma_f32_16x16x32_bf16(
                    false, af[i], false, bfr[j], (short)0, acc[i][j], false, false);
    };

    int nk = K >> 5;
    issue_tile(0);
    // steady state: branch-free; tile kk+1 in flight while tile kk computes
    for (int kk = 0; kk < nk - 1; kk++) {
        int buf = kk & 1;
        issue_tile(buf ^ 1);
        wait_async<NPEND>();   // tile kk landed; tile kk+1 may be in flight
        __syncthreads();
        compute_tile(buf);
        __syncthreads();       // all waves done reading buf before refill
    }
    // peeled last tile
    wait_async<0>();
    __syncthreads();
    compute_tile((nk - 1) & 1);

    // epilogue + store (C/D layout: VGPR v -> M = v + 8*(lane>>4), N = lane&15)
    int colL = lane & 15, rsel = (lane >> 4) << 3;
    #pragma unroll
    for (int i = 0; i < WTM; i++) {
        #pragma unroll
        for (int j = 0; j < WTN; j++) {
            int mB = m0 + wM + i * 16 + rsel;
            int nB = n0 + wN + j * 16 + colL;
            #pragma unroll
            for (int v = 0; v < 8; v++) {
                int row = mB + v;
                float val = acc[i][j][v];
                if (EPI == 1) val += p0[row];
                if (EPI == 2) val += p1[(size_t)bz * N + nB];
                if (EPI == 3) {
                    val += p0[((size_t)t * 64 + bz) * N + nB];
                    val = val > 0.f ? val : 0.f;
                }
                if (EPI == 4) val += p0[row] * p1[nB] + p2[row];
                size_t off = (size_t)bz * (size_t)Cbs + (size_t)row * N + nB;
                if (EPI == 2 || EPI == 4) ((float*)Cv)[off] = val;
                else                      ((u16*)Cv)[off]   = f2bf(val);
            }
        }
    }
}

// ------------------------------- launcher ----------------------------------

extern "C" void kernel_launch(void* const* d_in, const int* in_sizes, int n_in,
                              void* d_out, int out_size, void* d_ws, size_t ws_size,
                              hipStream_t stream) {
    const float* x   = (const float*)d_in[0];
    const float* W1  = (const float*)d_in[1];
    const float* b1  = (const float*)d_in[2];
    const float* Wl  = (const float*)d_in[3];
    const float* bl  = (const float*)d_in[4];
    const float* Wr  = (const float*)d_in[5];
    const float* Wf  = (const float*)d_in[6];
    const float* bfv = (const float*)d_in[7];
    const float* W2  = (const float*)d_in[8];
    const float* b2  = (const float*)d_in[9];
    float* out = (float*)d_out;

    char* w = (char*)d_ws;
    u16*   W1t    = (u16*)w;            w += (size_t)256 * 256 * 2;
    u16*   W2t    = (u16*)w;            w += (size_t)256 * 256 * 2;
    u16*   WfT    = (u16*)w;            w += (size_t)1024 * 512 * 2;
    u16*   WlCatT = (u16*)w;            w += (size_t)3 * 512 * 3072 * 2;
    u16*   WrSumT = (u16*)w;            w += (size_t)3 * 512 * 1024 * 2;
    float* blmean = (float*)w;          w += (size_t)3 * 512 * 4;
    float* s2     = (float*)w;          w += (size_t)256 * 4;
    u16*   xT     = (u16*)w;            w += (size_t)64 * 1024 * 256 * 2;
    u16*   H      = (u16*)w;            w += (size_t)64 * 256 * 1024 * 2;
    u16*   mucat  = (u16*)w;            w += (size_t)64 * 3072 * 2;
    float* alpha  = (float*)w;          w += (size_t)3 * 64 * 512 * 4;
    u16*   Z      = (u16*)w;            w += (size_t)64 * 256 * 512 * 2;
    u16*   Zt     = (u16*)w;            w += (size_t)64 * 512 * 256 * 2;
    u16*   G      = (u16*)w;            w += (size_t)64 * 256 * 512 * 2;

    // weight prep
    k_transpose_w<<<(256 * 256 + 255) / 256, 256, 0, stream>>>(W1, W1t, 256, 256);
    k_transpose_w<<<(256 * 256 + 255) / 256, 256, 0, stream>>>(W2, W2t, 256, 256);
    k_transpose_w<<<(512 * 1024 + 255) / 256, 256, 0, stream>>>(Wf, WfT, 512, 1024);
    k_wlcat<<<(3 * 512 * 3072 + 255) / 256, 256, 0, stream>>>(Wl, WlCatT, 3 * 512 * 3072);
    k_wrsum<<<(3 * 512 * 1024 + 255) / 256, 256, 0, stream>>>(Wr, WrSumT, 3 * 512 * 1024);
    k_blmean_s2<<<7, 256, 0, stream>>>(bl, W2, blmean, s2);

    // x -> xT (fp32 -> bf16 + transpose): [64][256][1024] -> [64][1024][256]
    k_transpose_tile<float><<<dim3(16, 4, 64), 256, 0, stream>>>(x, xT, 256, 1024);

    // H[b] = W1t @ xT[b]^T + b1     (M=256, N=1024, K=256)
    k_gemm<128, 128, 4, 2, 2, 4, 1, false><<<dim3(8, 2, 64), 256, 0, stream>>>(
        W1t, xT, H, 256, 1024, 256,
        0LL, 1024LL * 256, 256LL * 1024, b1, nullptr, nullptr);

    // slice means
    k_mu<<<dim3(3, 64), 256, 0, stream>>>(H, mucat);

    // alpha[t] = mucat @ WlCatT[t]^T + blmean[t]   (M=64, N=512, K=3072)
    k_gemm<64, 128, 2, 4, 2, 2, 2, false><<<dim3(4, 1, 3), 256, 0, stream>>>(
        mucat, WlCatT, alpha, 64, 512, 3072,
        0LL, 512LL * 3072, 64LL * 512, nullptr, blmean, nullptr);

    // Z[b] = relu(H[b] @ WrSumT[t]^T + alpha[t][b])  (M=256, N=512, K=1024)
    k_gemm<64, 128, 2, 4, 2, 2, 3, true><<<dim3(4, 4, 64), 256, 0, stream>>>(
        H, WrSumT, Z, 256, 512, 1024,
        256LL * 1024, 512LL * 1024, 256LL * 512, alpha, nullptr, nullptr);

    // Z -> Zt: [64][256][512] -> [64][512][256]
    k_transpose_tile<u16><<<dim3(8, 4, 64), 256, 0, stream>>>(Z, Zt, 256, 512);

    // G[b] = W2t @ Zt[b]^T          (M=256, N=512, K=256)
    k_gemm<128, 128, 4, 2, 2, 4, 0, false><<<dim3(4, 2, 64), 256, 0, stream>>>(
        W2t, Zt, G, 256, 512, 256,
        0LL, 512LL * 256, 256LL * 512, nullptr, nullptr, nullptr);

    // y[b] = G[b] @ WfT^T + s2[d]*bf[f] + b2[d]   (M=256, N=1024, K=512) -> f32 out
    k_gemm<128, 128, 4, 2, 2, 4, 4, false><<<dim3(8, 2, 64), 256, 0, stream>>>(
        G, WfT, out, 256, 1024, 512,
        256LL * 512, 0LL, 256LL * 1024, s2, bfv, b2);
}
